// FOFE_GRU_6287832121916
// MI455X (gfx1250) — matrix operand order repaired and checked by reading, not compiled, not run on hardware
//
#include <hip/hip_runtime.h>

typedef __attribute__((ext_vector_type(16))) _Float16 v16h;
typedef __attribute__((ext_vector_type(8)))  float    v8f;

#define B_  64
#define S_  512
#define W_  16
#define V_  1024
#define H_  256
#define L_  32
#define NG_ 768                     // 3*H

#define PACK_HALVES 196608          // 3 gates * 16 ctiles * 8 kchunks * 32 lanes * 16 halves
#define HFRAG 16384                 // 4 mtiles * 8 kchunks * 32 lanes * 16 halves

// ---------------------------------------------------------------------------
// Kernel 0a: pack w_hh (f32 [768,256]) into f16 WMMA B-fragment layout.
// Layout: [gate g][c_tile][k_chunk][lane][16 halves]; lane: col = lane&15,
// hi = lane>>4; half hidx -> K = kc*32 + (hidx>>3)*16 + hi*8 + (hidx&7); N = g*256+ct*16+col.
// ---------------------------------------------------------------------------
__global__ __launch_bounds__(256) void pack_whh_kernel(
    const float* __restrict__ whh_f, const float* __restrict__ whh_b,
    _Float16* __restrict__ packed) {
  int idx = blockIdx.x * 256 + threadIdx.x;            // < 2*PACK_HALVES
  int dir = idx >= PACK_HALVES;
  int e   = idx - dir * PACK_HALVES;
  int hidx = e & 15;
  int lane = (e >> 4) & 31;
  int kc   = (e >> 9) & 7;
  int rest = e >> 12;                                  // g*16 + ct
  int ct = rest & 15;
  int g  = rest >> 4;
  int n = g * H_ + ct * 16 + (lane & 15);
  int k = kc * 32 + (hidx >> 3) * 16 + (lane >> 4) * 8 + (hidx & 7);
  const float* w = dir ? whh_b : whh_f;
  packed[idx] = (_Float16)w[(size_t)n * H_ + k];
}

// ---------------------------------------------------------------------------
// Kernel 0b: rs[l] = sum_h w_lin[l,h]  (final output is rank-1 in H).
// ---------------------------------------------------------------------------
__global__ void rowsum_kernel(const float* __restrict__ w_lin,
                              float* __restrict__ rs) {
  int l = threadIdx.x;
  if (l < L_) {
    float a = 0.f;
    for (int h = 0; h < H_; ++h) a += w_lin[(size_t)l * H_ + h];
    rs[l] = a;
  }
}

// ---------------------------------------------------------------------------
// Kernel 1: FOFE + sparse input projection. One block per (b,s).
// gi[pos, g] = b_ih[g] + sum_k wt_k * w_ih[g, tok_k], both directions.
// ---------------------------------------------------------------------------
__global__ __launch_bounds__(256) void fofe_gates_kernel(
    const int* __restrict__ sents, const float* __restrict__ alpha,
    const float* __restrict__ wih_f, const float* __restrict__ bih_f,
    const float* __restrict__ wih_b, const float* __restrict__ bih_b,
    float* __restrict__ gi_f, float* __restrict__ gi_b) {
  __shared__ int   st[W_];
  __shared__ float swt[W_];
  const int pos = blockIdx.x;                 // b*S + s
  const int tid = threadIdx.x;
  if (tid == 0) {
    const float a = alpha[0];
    float pw[W_ + 1];
    pw[0] = 1.f;
    for (int i = 1; i <= W_; ++i) pw[i] = pw[i - 1] * a;
    int toks[W_], cl[W_], cnt = 0;
    for (int k = 0; k < W_; ++k) {
      int t = sents[(size_t)pos * W_ + k];
      toks[k] = t;
      cnt += (t != 0);
      cl[k] = cnt;                            // inclusive count of nonzeros
    }
    for (int k = 0; k < W_; ++k) {
      st[k]  = toks[k];
      swt[k] = (toks[k] != 0) ? pw[cnt - cl[k]] : 0.f;
    }
  }
  __syncthreads();
  const int c = tid;                          // hidden channel 0..255
  const size_t gbase = (size_t)pos * NG_;
#pragma unroll
  for (int g = 0; g < 3; ++g) {
    int ch = g * H_ + c;
    float af = bih_f[ch], ab = bih_b[ch];
    const float* rf = wih_f + (size_t)ch * V_;
    const float* rb = wih_b + (size_t)ch * V_;
#pragma unroll
    for (int k = 0; k < W_; ++k) {
      float w = swt[k];
      int   t = st[k];
      af += w * rf[t];
      ab += w * rb[t];
    }
    gi_f[gbase + ch] = af;
    gi_b[gbase + ch] = ab;
  }
}

// ---------------------------------------------------------------------------
// Kernel 2: GRU recurrence. blockIdx.x = direction. 512 threads = 16 waves.
// Each wave owns 4 (m_tile, c_tile) units; per unit it computes the r/z/n
// 16x16 gh tiles via v_wmma_f32_16x16x32_f16 (K=256 -> 8 chunks), then does
// the GRU gate math in registers and writes the updated h directly into a
// double-buffered f16 A-fragment image in LDS (one barrier per timestep).
// The unit loop is kept a runtime loop (unroll 1) so the loop-invariant
// weight fragments are NOT hoisted across the t-loop (which previously
// spilled ~768 VGPRs of w_hh to scratch).
// ---------------------------------------------------------------------------
__global__ __launch_bounds__(512) void gru_kernel(
    const float* __restrict__ gi_f, const float* __restrict__ gi_b,
    const _Float16* __restrict__ packed,
    const float* __restrict__ bhh_f, const float* __restrict__ bhh_b,
    const int* __restrict__ lengths,
    float* __restrict__ val_f, float* __restrict__ val_r) {
  const int dir = blockIdx.x;
  const float*    gi  = dir ? gi_b  : gi_f;
  const float*    bhh = dir ? bhh_b : bhh_f;
  const _Float16* pk  = packed + (size_t)dir * PACK_HALVES;
  float* valout = dir ? val_r : val_f;

  __shared__ _Float16 hbuf[2][HFRAG] __attribute__((aligned(32))); // 2x32KB
  __shared__ int s_len[B_];
  __shared__ int s_sel[B_];

  const int tid = threadIdx.x;
  for (int i = tid; i < HFRAG; i += 512) hbuf[0][i] = (_Float16)0.f;
  if (tid < B_) {
    int len = lengths[tid];
    s_len[tid] = len;
    int idxb = len - 1;                       // index into 2H concat
    s_sel[tid] = dir ? ((idxb >= H_) ? (idxb - H_) : -1)
                     : ((idxb <  H_) ?  idxb       : -1);
  }
  __syncthreads();

  const int wave = tid >> 5;
  const int lane = tid & 31;
  const int col  = lane & 15;
  const int hi   = lane >> 4;

  for (int t = 0; t < S_; ++t) {
    const _Float16* hr = hbuf[t & 1];         // read image
    _Float16*       hw = hbuf[(t & 1) ^ 1];   // write image (next step's read)

    // prefetch next timestep's gate-preactivation slice (global_prefetch_b8)
    {
      int tn = (t + 1 < S_) ? (t + 1) : t;
      int mpf = tid >> 3;                     // 0..63
      int srow = tn;
      if (dir) {
        srow = s_len[mpf] - 1 - tn;
        srow = srow < 0 ? 0 : srow;
      }
      const float* pf = gi + ((size_t)mpf * S_ + srow) * NG_ + (tid & 7) * 96;
      __builtin_prefetch(pf, 0, 0);
    }

#pragma unroll 1
    for (int j = 0; j < 4; ++j) {
      const int u  = wave * 4 + j;
      const int mt = u & 3;
      const int ct = u >> 2;
      v8f accR = {}, accZ = {}, accN = {};
#pragma unroll
      for (int kc = 0; kc < 8; ++kc) {
        v16h a = *(const v16h*)&hr[((mt * 8 + kc) * 32 + lane) * 16];
        const _Float16* pb = pk + ((size_t)((ct * 8 + kc) * 32 + lane)) * 16;
        v16h br = *(const v16h*)(pb);
        v16h bz = *(const v16h*)(pb + 65536);          // gate stride in halves
        v16h bn = *(const v16h*)(pb + 131072);
        accR = __builtin_amdgcn_wmma_f32_16x16x32_f16(false, a, false, br,
                                                      (short)0, accR, false, false);
        accZ = __builtin_amdgcn_wmma_f32_16x16x32_f16(false, a, false, bz,
                                                      (short)0, accZ, false, false);
        accN = __builtin_amdgcn_wmma_f32_16x16x32_f16(false, a, false, bn,
                                                      (short)0, accN, false, false);
      }
      const int   c   = ct * 16 + col;
      const float br_ = bhh[c], bz_ = bhh[H_ + c], bn_ = bhh[2 * H_ + c];
      // inverse A-fragment mapping for channel c (h_prev lookup / h write)
      const int kc2 = c >> 5, ko = c & 31;
      const int hi2 = (ko >> 3) & 1;
      const int w2  = ko - 8 * hi2;
      const int p2  = (w2 < 8) ? (w2 >> 1) : (4 + ((w2 - 16) >> 1));
      const int hidx = 2 * p2 + (w2 & 1);
#pragma unroll
      for (int r = 0; r < 8; ++r) {
        const int m   = mt * 16 + r + 8 * hi;
        const int len = s_len[m];
        int srow = t;
        if (dir) { srow = len - 1 - t; srow = srow < 0 ? 0 : srow; }
        const size_t gp = ((size_t)m * S_ + srow) * NG_;
        const float gr = gi[gp + c], gz = gi[gp + H_ + c], gn = gi[gp + 2 * H_ + c];
        const float ghr = accR[r] + br_;
        const float ghz = accZ[r] + bz_;
        const float ghn = accN[r] + bn_;
        const float rg = 1.f / (1.f + __expf(-(gr + ghr)));
        const float zg = 1.f / (1.f + __expf(-(gz + ghz)));
        const float ng = tanhf(gn + rg * ghn);
        const int   hoff =
            ((mt * 8 + kc2) * 32 + (hi2 * 16 + (m & 15))) * 16 + hidx;
        const float hprev = (float)hr[hoff];
        const float hv  = (1.f - zg) * ng + zg * hprev;
        const bool  msk = t < len;
        hw[hoff] = (_Float16)(msk ? hv : hprev);
        if (s_sel[m] == c) valout[(size_t)m * S_ + t] = msk ? hv : 0.f;
      }
    }
    __syncthreads();                      // h(t+1) image complete
  }
}

// ---------------------------------------------------------------------------
// Kernel 3: out[b,s,l] = val[b,s] * rowsum(w_lin[l]) + b_lin[l]
// (reshape(B,L,S) is a pure memory reinterpretation of flat [B,S,L]).
// ---------------------------------------------------------------------------
__global__ __launch_bounds__(256) void out_kernel(
    const int* __restrict__ lengths, const float* __restrict__ val_f,
    const float* __restrict__ val_r, const float* __restrict__ rs,
    const float* __restrict__ b_lin, float* __restrict__ out) {
  const int i = blockIdx.x * 256 + threadIdx.x;   // < B*S*L
  const int l = i & (L_ - 1);
  const int s = (i >> 5) & (S_ - 1);
  const int b = i >> 14;
  const int len  = lengths[b];
  const int idxb = len - 1;
  float v;
  if (idxb < H_) v = val_f[(size_t)b * S_ + s];
  else           v = (s < len) ? val_r[(size_t)b * S_ + (len - 1 - s)] : 0.f;
  out[i] = v * rs[l] + b_lin[l];
}

// ---------------------------------------------------------------------------
extern "C" void kernel_launch(void* const* d_in, const int* in_sizes, int n_in,
                              void* d_out, int out_size, void* d_ws, size_t ws_size,
                              hipStream_t stream) {
  const int*   sents   = (const int*)d_in[0];
  const int*   lengths = (const int*)d_in[1];
  const float* alpha   = (const float*)d_in[2];
  const float* w_ih_f  = (const float*)d_in[3];
  const float* w_hh_f  = (const float*)d_in[4];
  const float* b_ih_f  = (const float*)d_in[5];
  const float* b_hh_f  = (const float*)d_in[6];
  const float* w_ih_b  = (const float*)d_in[7];
  const float* w_hh_b  = (const float*)d_in[8];
  const float* b_ih_b  = (const float*)d_in[9];
  const float* b_hh_b  = (const float*)d_in[10];
  const float* w_lin   = (const float*)d_in[11];
  const float* b_lin   = (const float*)d_in[12];
  float* out = (float*)d_out;

  // workspace layout (bytes)
  char* ws = (char*)d_ws;
  const size_t OFF_PACK  = 0;                                   // 786432 B
  const size_t OFF_GI_F  = 786432;                              // 100663296 B
  const size_t OFF_GI_B  = OFF_GI_F + 100663296ull;
  const size_t OFF_VAL_F = OFF_GI_B + 100663296ull;             // 131072 B
  const size_t OFF_VAL_R = OFF_VAL_F + 131072ull;
  const size_t OFF_RS    = OFF_VAL_R + 131072ull;

  _Float16* packed = (_Float16*)(ws + OFF_PACK);
  float* gi_f  = (float*)(ws + OFF_GI_F);
  float* gi_b  = (float*)(ws + OFF_GI_B);
  float* val_f = (float*)(ws + OFF_VAL_F);
  float* val_r = (float*)(ws + OFF_VAL_R);
  float* rs    = (float*)(ws + OFF_RS);

  pack_whh_kernel<<<(2 * PACK_HALVES) / 256, 256, 0, stream>>>(w_hh_f, w_hh_b,
                                                               packed);
  rowsum_kernel<<<1, 64, 0, stream>>>(w_lin, rs);
  fofe_gates_kernel<<<B_ * S_, 256, 0, stream>>>(sents, alpha, w_ih_f, b_ih_f,
                                                 w_ih_b, b_ih_b, gi_f, gi_b);
  gru_kernel<<<2, 512, 0, stream>>>(gi_f, gi_b, packed, b_hh_f, b_hh_b,
                                    lengths, val_f, val_r);
  out_kernel<<<(B_ * S_ * L_) / 256, 256, 0, stream>>>(lengths, val_f, val_r,
                                                       rs, b_lin, out);
}
